// PWBLinearLayer_68985764708477
// MI455X (gfx1250) — compile-verified, matrix operand order checked by reading, not run on hardware
//
#include <hip/hip_runtime.h>

typedef __attribute__((ext_vector_type(16))) _Float16 v16h;
typedef __attribute__((ext_vector_type(8)))  _Float16 v8h;
typedef __attribute__((ext_vector_type(8)))  float    v8f;
typedef __attribute__((ext_vector_type(4)))  float    v4f;

#define D_IN   1024
#define D_OUT  128
#define KCHUNK 128              // K-elements staged in LDS per phase
#define LDS_STRIDE 136          // 128 + 8 halves pad (16B-aligned rows, staggers banks)

// One-shot: quantize W (round(clip*127) is an integer, exact in fp16) and
// transpose to Wt[n][k] so the GEMM stages LDS with pure b128 copies.
__global__ __launch_bounds__(256)
void pwb_quant_transpose_kernel(const float* __restrict__ W,
                                _Float16* __restrict__ Wt) {
    const int idx = blockIdx.x * blockDim.x + threadIdx.x;  // 16384 threads
    const int n = idx >> 7;            // 0..127
    const int k = (idx & 127) * 8;     // 0..1016 step 8
    v8h q;
    #pragma unroll
    for (int j = 0; j < 8; ++j) {
        float w = W[(size_t)(k + j) * D_OUT + n];
        w = fminf(fmaxf(w, -1.0f), 1.0f) * 127.0f;
        q[j] = (_Float16)rintf(w);
    }
    *(v8h*)(Wt + (size_t)n * D_IN + k) = q;
}

template <bool PREQ>
__global__ __launch_bounds__(128)
void pwb_linear_wmma_kernel(const float* __restrict__ x,
                            const void*  __restrict__ Wsrc,
                            const float* __restrict__ b,
                            float* __restrict__ out) {
    __shared__ _Float16 ldsW[D_OUT * LDS_STRIDE];   // 34816 B

    const int tid  = threadIdx.x;
    const int wave = tid >> 5;          // 4 waves per block
    const int lane = tid & 31;
    const int ln   = lane & 15;
    const int hi   = lane >> 4;
    const int koff = hi * 8;            // ISA 16-bit A/B lane K offset

    const int m0 = blockIdx.x * 128 + wave * 32;    // 32-row strip per wave

    v8f acc0[8] = {};                   // rows m0..m0+15
    v8f acc1[8] = {};                   // rows m0+16..m0+31

    const float* xrow0 = x + (size_t)(m0 + ln) * D_IN;
    const float* xrow1 = xrow0 + (size_t)16 * D_IN;

    for (int kc = 0; kc < D_IN / KCHUNK; ++kc) {
        const int kbase = kc * KCHUNK;

        __syncthreads();    // previous chunk's LDS reads complete
        if (PREQ) {
            // Coalesced b128 copy of pre-quantized transposed chunk into LDS
            const _Float16* Wt = (const _Float16*)Wsrc;
            for (int i = tid; i < D_OUT * (KCHUNK / 8); i += 128) {
                const int n = i >> 4;          // 0..127
                const int g = (i & 15) * 8;    // 0..120 step 8
                v8h v = *(const v8h*)(Wt + (size_t)n * D_IN + kbase + g);
                *(v8h*)(&ldsW[n * LDS_STRIDE + g]) = v;
            }
        } else {
            // Fallback: quantize+transpose inline (workspace too small)
            const float* W = (const float*)Wsrc;
            for (int i = tid; i < KCHUNK * D_OUT; i += 128) {
                const int n  = i & (D_OUT - 1);
                const int kk = i >> 7;
                float w = W[(size_t)(kbase + kk) * D_OUT + n];
                w = fminf(fmaxf(w, -1.0f), 1.0f) * 127.0f;
                ldsW[n * LDS_STRIDE + kk] = (_Float16)rintf(w);
            }
        }
        __syncthreads();

        if (kc + 1 < D_IN / KCHUNK) {
            __builtin_prefetch(xrow0 + kbase + KCHUNK + koff, 0, 0);
            __builtin_prefetch(xrow1 + kbase + KCHUNK + koff, 0, 0);
        }

        #pragma unroll
        for (int ks = 0; ks < KCHUNK / 32; ++ks) {
            const int k0 = kbase + ks * 32;

            // Two A fragments (16x32 f16 each): lane holds one row,
            // K = k0+koff+{0..7} and k0+koff+{16..23}
            const float* ap0 = xrow0 + k0 + koff;
            const float* ap1 = xrow1 + k0 + koff;
            v4f a00 = *(const v4f*)(ap0 + 0);
            v4f a01 = *(const v4f*)(ap0 + 4);
            v4f a02 = *(const v4f*)(ap0 + 16);
            v4f a03 = *(const v4f*)(ap0 + 20);
            v4f a10 = *(const v4f*)(ap1 + 0);
            v4f a11 = *(const v4f*)(ap1 + 4);
            v4f a12 = *(const v4f*)(ap1 + 16);
            v4f a13 = *(const v4f*)(ap1 + 20);
            v16h af0, af1;
            #pragma unroll
            for (int i = 0; i < 4; ++i) {
                af0[i]      = (_Float16)a00[i];
                af0[4 + i]  = (_Float16)a01[i];
                af0[8 + i]  = (_Float16)a02[i];
                af0[12 + i] = (_Float16)a03[i];
                af1[i]      = (_Float16)a10[i];
                af1[4 + i]  = (_Float16)a11[i];
                af1[8 + i]  = (_Float16)a12[i];
                af1[12 + i] = (_Float16)a13[i];
            }

            // 8 N-tiles: each B fragment feeds TWO WMMAs (A reuse across M)
            #pragma unroll
            for (int nt = 0; nt < 8; ++nt) {
                const _Float16* bp =
                    &ldsW[(nt * 16 + ln) * LDS_STRIDE + ks * 32 + koff];
                v8h blo = *(const v8h*)(bp);        // K koff+0..7
                v8h bhi = *(const v8h*)(bp + 16);   // K koff+16..23
                v16h bf;
                #pragma unroll
                for (int i = 0; i < 8; ++i) {
                    bf[i]     = blo[i];
                    bf[8 + i] = bhi[i];
                }
                acc0[nt] = __builtin_amdgcn_wmma_f32_16x16x32_f16(
                    false, af0, false, bf, (short)0, acc0[nt], false, false);
                acc1[nt] = __builtin_amdgcn_wmma_f32_16x16x32_f16(
                    false, af1, false, bf, (short)0, acc1[nt], false, false);
            }
        }
    }

    // Epilogue: D = acc/127 + quantized bias
    const float scale = 1.0f / 127.0f;
    #pragma unroll
    for (int nt = 0; nt < 8; ++nt) {
        const int col = nt * 16 + ln;
        float bv = b[col];
        bv = rintf(fminf(fmaxf(bv, -1.0f), 1.0f) * 127.0f) * scale;
        const int r0 = m0 + hi * 8;         // strip 0
        const int r1 = m0 + 16 + hi * 8;    // strip 1
        #pragma unroll
        for (int r = 0; r < 8; ++r) {
            out[(size_t)(r0 + r) * D_OUT + col] = acc0[nt][r] * scale + bv;
            out[(size_t)(r1 + r) * D_OUT + col] = acc1[nt][r] * scale + bv;
        }
    }
}

extern "C" void kernel_launch(void* const* d_in, const int* in_sizes, int n_in,
                              void* d_out, int out_size, void* d_ws, size_t ws_size,
                              hipStream_t stream) {
    const float* x = (const float*)d_in[0];   // [B, 1024]
    const float* W = (const float*)d_in[1];   // [1024, 128]
    const float* b = (const float*)d_in[2];   // [128]
    float* out = (float*)d_out;               // [B, 128]

    const int B = in_sizes[0] / D_IN;         // 32768
    dim3 grid(B / 128);                       // 128 rows per block (4 waves x 32)
    dim3 block(128);

    const size_t wt_bytes = (size_t)D_IN * D_OUT * sizeof(_Float16);  // 256 KB
    if (ws_size >= wt_bytes) {
        _Float16* Wt = (_Float16*)d_ws;
        pwb_quant_transpose_kernel<<<dim3(64), dim3(256), 0, stream>>>(W, Wt);
        pwb_linear_wmma_kernel<true><<<grid, block, 0, stream>>>(x, Wt, b, out);
    } else {
        pwb_linear_wmma_kernel<false><<<grid, block, 0, stream>>>(x, W, b, out);
    }
}
